// LidarGatLstmNet_20779051778367
// MI455X (gfx1250) — matrix operand order repaired and checked by reading, not compiled, not run on hardware
//
#include <hip/hip_runtime.h>
#include <hip/hip_bf16.h>

typedef __attribute__((ext_vector_type(16))) _Float16 v16h;
typedef __attribute__((ext_vector_type(8)))  _Float16 v8h;
typedef __attribute__((ext_vector_type(8)))  float    v8f;

#define NN   8640
#define NPG  1080
#define BB   8
#define DEG  16

// ---------------- Layer 1: K=2 GEMM (VALU) ----------------
__global__ void lin1_kernel(const float* __restrict__ x, const float* __restrict__ W,
                            float* __restrict__ h, int Nc) {
    int idx = blockIdx.x * blockDim.x + threadIdx.x;
    if (idx >= NN * Nc) return;
    int n = idx / Nc;
    int j = idx - n * Nc;
    h[idx] = x[n * 2] * W[j] + x[n * 2 + 1] * W[Nc + j];
}

// ---------------- Small-N GEMM (layer 4, N=8) ----------------
__global__ void lin_small_kernel(const float* __restrict__ x, const float* __restrict__ W,
                                 float* __restrict__ h, int K, int Nc) {
    int idx = blockIdx.x * blockDim.x + threadIdx.x;
    if (idx >= NN * Nc) return;
    int n = idx / Nc;
    int j = idx - n * Nc;
    float s = 0.f;
    for (int k = 0; k < K; k++) s += x[(size_t)n * K + k] * W[k * Nc + j];
    h[idx] = s;
}

// ---------------- Pack W [K,Nc] f32 -> WMMA B-fragment order f16 ----------------
// packed[((kt*ctiles+ct)*32 + lane)*16 + i] = W[kt*32 + (i>=8?16:0) + (lane>=16?8:0) + (i&7)][ct*16 + (lane&15)]
__global__ void pack_b_kernel(const float* __restrict__ W, _Float16* __restrict__ out,
                              int K, int Nc) {
    int tid = blockIdx.x * blockDim.x + threadIdx.x;
    if (tid >= K * Nc) return;
    int tile = tid >> 9;
    int r    = tid & 511;
    int l    = r >> 4;
    int i    = r & 15;
    int ctiles = Nc >> 4;
    int kt = tile / ctiles;
    int ct = tile - kt * ctiles;
    int k   = kt * 32 + ((i >= 8) ? 16 : 0) + ((l >= 16) ? 8 : 0) + (i & 7);
    int col = ct * 16 + (l & 15);
    out[tid] = (_Float16)W[k * Nc + col];
}

// ---------------- WMMA GEMM: [NN x K](f16 rowmajor) x packed-B -> C f32 ----------------
__global__ void gemm_wmma_f16(const _Float16* __restrict__ A, const _Float16* __restrict__ Bp,
                              float* __restrict__ C, int K, int Nc) {
    int lane = threadIdx.x;
    int row0 = blockIdx.x << 4;
    int ct   = blockIdx.y;
    int ctiles = Nc >> 4;
    int arow = row0 + (lane & 15);
    int asel = (lane >> 4) << 3;               // 0 or 8
    const _Float16* Ar = A + (size_t)arow * K + asel;
    const _Float16* Bt = Bp + (size_t)ct * 512 + lane * 16;
    v8f acc = {};
    int ktiles = K >> 5;
    for (int kt = 0; kt < ktiles; kt++) {
        v8h a0 = *(const v8h*)(Ar + kt * 32);
        v8h a1 = *(const v8h*)(Ar + kt * 32 + 16);
        const _Float16* Bb = Bt + (size_t)kt * ctiles * 512;
        v8h b0 = *(const v8h*)(Bb);
        v8h b1 = *(const v8h*)(Bb + 8);
        v16h a, b;
#pragma unroll
        for (int i = 0; i < 8; i++) { a[i] = a0[i]; a[i + 8] = a1[i]; b[i] = b0[i]; b[i + 8] = b1[i]; }
        acc = __builtin_amdgcn_wmma_f32_16x16x32_f16(false, a, false, b, (short)0, acc, false, false);
    }
    int col   = (ct << 4) + (lane & 15);
    int rbase = row0 + ((lane >> 4) << 3);
#pragma unroll
    for (int r = 0; r < 8; r++) C[(size_t)(rbase + r) * Nc + col] = acc[r];
}

// ---------------- Per-node/head attention logits: es = <h, a_s>, ed = <h, a_d> ----------------
__global__ void gat_edge_kernel(const float* __restrict__ h, const float* __restrict__ as_,
                                const float* __restrict__ ad_, float* __restrict__ es,
                                float* __restrict__ ed, int Hh, int C) {
    int idx = blockIdx.x * blockDim.x + threadIdx.x;
    if (idx >= NN * Hh) return;
    int n  = idx / Hh;
    int hh = idx - n * Hh;
    const float* hp  = h + (size_t)n * Hh * C + hh * C;
    const float* asp = as_ + hh * C;
    const float* adp = ad_ + hh * C;
    float ss = 0.f, sd = 0.f;
    for (int c = 0; c < C; c++) { float v = hp[c]; ss += v * asp[c]; sd += v * adp[c]; }
    es[idx] = ss;
    ed[idx] = sd;
}

// ---------------- Attention softmax + aggregation + bias + ELU ----------------
// Each dst node n receives edges src[n*16 .. n*16+15] plus self-loop (by construction).
__global__ void gat_aggr_kernel(const float* __restrict__ h, const float* __restrict__ es,
                                const float* __restrict__ ed, const float* __restrict__ bias,
                                const int* __restrict__ src, float* __restrict__ yout,
                                _Float16* __restrict__ y16, int Hh, int C, int writeHalf) {
    int HC = Hh * C;
    int idx = blockIdx.x * blockDim.x + threadIdx.x;
    if (idx >= NN * HC) return;
    int n   = idx / HC;
    int rem = idx - n * HC;
    int hh  = rem / C;
    float edn = ed[n * Hh + hh];
    int s[17];
    const int* sp = src + n * DEG;
#pragma unroll
    for (int k = 0; k < DEG; k++) s[k] = sp[k];
    s[16] = n;
    float ev[17];
    float m = -3.4e38f;
#pragma unroll
    for (int k = 0; k < 17; k++) {
        float e = es[s[k] * Hh + hh] + edn;
        e = (e > 0.f) ? e : 0.2f * e;          // leaky_relu 0.2
        ev[k] = e;
        m = fmaxf(m, e);
    }
    float den = 0.f, acc = 0.f;
#pragma unroll
    for (int k = 0; k < 17; k++) {
        float w = __expf(ev[k] - m);
        den += w;
        acc += w * h[(size_t)s[k] * HC + rem];
    }
    float y = acc / (den + 1e-16f) + bias[rem];
    y = (y > 0.f) ? y : (__expf(y) - 1.f);     // ELU
    yout[idx] = y;
    if (writeHalf) y16[idx] = (_Float16)y;
}

// ---------------- Mean pool per graph: [NN,8] -> [B,8] ----------------
__global__ void pool_kernel(const float* __restrict__ y, float* __restrict__ emb_t) {
    int tid = threadIdx.x;                     // 64 threads: (b, c)
    int b = tid >> 3, c = tid & 7;
    const float* p = y + (size_t)b * NPG * 8 + c;
    float s = 0.f;
    for (int i = 0; i < NPG; i++) s += p[i * 8];
    emb_t[tid] = s * (1.0f / (float)NPG);
}

// ---------------- LSTM (T=10, B=8, H=128, in=8) + FC ----------------
__global__ void lstm_fc_kernel(const float* __restrict__ emb, const float* __restrict__ w_ih,
                               const float* __restrict__ w_hh, const float* __restrict__ b_ih,
                               const float* __restrict__ b_hh, const float* __restrict__ w_fc,
                               const float* __restrict__ b_fc, float* __restrict__ out) {
    __shared__ float hs[8 * 128];
    __shared__ float cs[8 * 128];
    int tid = threadIdx.x;                     // 1024 threads: b = tid>>7, j = tid&127
    int b = tid >> 7, j = tid & 127;
    hs[tid] = 0.f; cs[tid] = 0.f;
    __syncthreads();
    for (int t = 0; t < 10; t++) {
        const float* xt = emb + t * 64 + b * 8;
        const float* hb = hs + b * 128;
        float g4[4];
#pragma unroll
        for (int gi = 0; gi < 4; gi++) {
            int r = gi * 128 + j;
            float s = b_ih[r] + b_hh[r];
            const float* wi = w_ih + r * 8;
#pragma unroll
            for (int k = 0; k < 8; k++) s += xt[k] * wi[k];
            const float* wh = w_hh + r * 128;
            for (int k = 0; k < 128; k++) s += hb[k] * wh[k];
            g4[gi] = s;
        }
        float ig = 1.f / (1.f + __expf(-g4[0]));
        float fg = 1.f / (1.f + __expf(-g4[1]));
        float gg = tanhf(g4[2]);
        float og = 1.f / (1.f + __expf(-g4[3]));
        float cn = fg * cs[tid] + ig * gg;
        float hn = og * tanhf(cn);
        __syncthreads();
        hs[tid] = hn; cs[tid] = cn;
        __syncthreads();
    }
    if (tid < 16) {
        int bb = tid >> 1, o = tid & 1;
        float s = b_fc[o];
        const float* wf = w_fc + o * 128;
        const float* hb = hs + bb * 128;
        for (int k = 0; k < 128; k++) s += hb[k] * wf[k];
        out[bb * 2 + o] = s;
    }
}

static inline int cdiv(int a, int b) { return (a + b - 1) / b; }

extern "C" void kernel_launch(void* const* d_in, const int* in_sizes, int n_in,
                              void* d_out, int out_size, void* d_ws, size_t ws_size,
                              hipStream_t stream) {
    (void)in_sizes; (void)n_in; (void)out_size; (void)ws_size;
    const float* x_seq = (const float*)d_in[0];
    const int*   src   = (const int*)d_in[1];       // edge_index row 0 = src, E ints
    const float* W1 = (const float*)d_in[3];
    const float* as1 = (const float*)d_in[4], *ad1 = (const float*)d_in[5], *b1 = (const float*)d_in[6];
    const float* W2 = (const float*)d_in[7];
    const float* as2 = (const float*)d_in[8], *ad2 = (const float*)d_in[9], *b2 = (const float*)d_in[10];
    const float* W3 = (const float*)d_in[11];
    const float* as3 = (const float*)d_in[12], *ad3 = (const float*)d_in[13], *b3 = (const float*)d_in[14];
    const float* W4 = (const float*)d_in[15];
    const float* as4 = (const float*)d_in[16], *ad4 = (const float*)d_in[17], *b4 = (const float*)d_in[18];
    const float* w_ih = (const float*)d_in[19], *w_hh = (const float*)d_in[20];
    const float* b_ih = (const float*)d_in[21], *b_hh = (const float*)d_in[22];
    const float* w_fc = (const float*)d_in[23], *b_fc = (const float*)d_in[24];

    // Workspace carve-up (all 256B-aligned)
    char* ws = (char*)d_ws;
    size_t offH   = 0;                                  // [NN,512] f32
    size_t offY   = offH   + (size_t)NN * 512 * 4;      // [NN,512] f32
    size_t offX16 = offY   + (size_t)NN * 512 * 4;      // [NN,512] f16
    size_t offES  = offX16 + (size_t)NN * 512 * 2;      // [NN,8]   f32
    size_t offED  = offES  + (size_t)NN * 8 * 4;        // [NN,8]   f32
    size_t offW2  = offED  + (size_t)NN * 8 * 4;        // 512*256 f16 packed
    size_t offW3  = offW2  + (size_t)512 * 256 * 2;     // 256*128 f16 packed
    size_t offEMB = offW3  + (size_t)256 * 128 * 2;     // [10,8,8] f32

    float*    Hbuf = (float*)(ws + offH);
    float*    Ybuf = (float*)(ws + offY);
    _Float16* X16  = (_Float16*)(ws + offX16);
    float*    ES   = (float*)(ws + offES);
    float*    ED   = (float*)(ws + offED);
    _Float16* W2p  = (_Float16*)(ws + offW2);
    _Float16* W3p  = (_Float16*)(ws + offW3);
    float*    EMB  = (float*)(ws + offEMB);

    // Pack W2 / W3 to WMMA B-fragment layout (f16)
    pack_b_kernel<<<cdiv(512 * 256, 256), 256, 0, stream>>>(W2, W2p, 512, 256);
    pack_b_kernel<<<cdiv(256 * 128, 256), 256, 0, stream>>>(W3, W3p, 256, 128);

    for (int t = 0; t < 10; t++) {
        const float* x0 = x_seq + (size_t)t * NN * 2;

        // ---- Layer 1: in=2, H=8, C=64 (HC=512) ----
        lin1_kernel<<<cdiv(NN * 512, 256), 256, 0, stream>>>(x0, W1, Hbuf, 512);
        gat_edge_kernel<<<cdiv(NN * 8, 256), 256, 0, stream>>>(Hbuf, as1, ad1, ES, ED, 8, 64);
        gat_aggr_kernel<<<cdiv(NN * 512, 256), 256, 0, stream>>>(Hbuf, ES, ED, b1, src, Ybuf, X16, 8, 64, 1);

        // ---- Layer 2: K=512 -> HC=256 (WMMA) ----
        gemm_wmma_f16<<<dim3(NN / 16, 256 / 16), 32, 0, stream>>>(X16, W2p, Hbuf, 512, 256);
        gat_edge_kernel<<<cdiv(NN * 8, 256), 256, 0, stream>>>(Hbuf, as2, ad2, ES, ED, 8, 32);
        gat_aggr_kernel<<<cdiv(NN * 256, 256), 256, 0, stream>>>(Hbuf, ES, ED, b2, src, Ybuf, X16, 8, 32, 1);

        // ---- Layer 3: K=256 -> HC=128 (WMMA) ----
        gemm_wmma_f16<<<dim3(NN / 16, 128 / 16), 32, 0, stream>>>(X16, W3p, Hbuf, 256, 128);
        gat_edge_kernel<<<cdiv(NN * 8, 256), 256, 0, stream>>>(Hbuf, as3, ad3, ES, ED, 8, 16);
        gat_aggr_kernel<<<cdiv(NN * 128, 256), 256, 0, stream>>>(Hbuf, ES, ED, b3, src, Ybuf, X16, 8, 16, 0);

        // ---- Layer 4: K=128 -> HC=8 (VALU, N too small for WMMA tile) ----
        lin_small_kernel<<<cdiv(NN * 8, 256), 256, 0, stream>>>(Ybuf, W4, Hbuf, 128, 8);
        gat_edge_kernel<<<cdiv(NN * 1, 256), 256, 0, stream>>>(Hbuf, as4, ad4, ES, ED, 1, 8);
        gat_aggr_kernel<<<cdiv(NN * 8, 256), 256, 0, stream>>>(Hbuf, ES, ED, b4, src, Ybuf, X16, 1, 8, 0);

        // ---- Mean pool per graph -> emb[t] ----
        pool_kernel<<<1, 64, 0, stream>>>(Ybuf, EMB + t * 64);
    }

    // ---- LSTM over 10 steps + FC ----
    lstm_fc_kernel<<<1, 1024, 0, stream>>>(EMB, w_ih, w_hh, b_ih, b_hh, w_fc, b_fc, (float*)d_out);
}
